// CausalSelfAttention_4415226380937
// MI455X (gfx1250) — matrix operand order repaired and checked by reading, not compiled, not run on hardware
//
#include <hip/hip_runtime.h>

#define Bb 4
#define Hh 4
#define Ss 4096
#define Ee 64
#define Dd 16
#define Dp 32   // head dim zero-padded to 32 (WMMA f16 K dimension)

typedef __attribute__((ext_vector_type(16))) _Float16 v16h;
typedef __attribute__((ext_vector_type(8)))  _Float16 v8h;
typedef __attribute__((ext_vector_type(8)))  float    v8f;
typedef __attribute__((ext_vector_type(4)))  unsigned v4u;
typedef __attribute__((ext_vector_type(8)))  int      v8i;
typedef __attribute__((ext_vector_type(4)))  int      v4i;

#if __has_builtin(__builtin_amdgcn_tensor_load_to_lds)
#define USE_TDM 1
#else
#define USE_TDM 0
#endif

static __device__ __forceinline__ v8f wmma_f16(v16h a, v16h b, v8f c) {
    return __builtin_amdgcn_wmma_f32_16x16x32_f16(false, a, false, b, (short)0, c, false, false);
}

static __device__ __forceinline__ v16h cat8(v8h lo, v8h hi) {
    return __builtin_shufflevector(lo, hi, 0,1,2,3,4,5,6,7,8,9,10,11,12,13,14,15);
}

static __device__ __forceinline__ float dot64(const float* __restrict__ a,
                                              const float* __restrict__ b) {
    float acc = 0.f;
#pragma unroll
    for (int e = 0; e < Ee; e += 4) {
        float4 av = *(const float4*)(a + e);
        float4 bv = *(const float4*)(b + e);
        acc += av.x * bv.x + av.y * bv.y + av.z * bv.z + av.w * bv.w;
    }
    return acc;
}

// ---------------------------------------------------------------------------
// Kernel 1: fused QKV projection + RoPE.
//   qh, kh : [B][H][S][Dp] f16 (head dim padded 16->32 with zeros)
//   vt     : [B][H][Dd][S] f16 (transposed: PV B-fragment becomes contiguous)
// ---------------------------------------------------------------------------
__global__ __launch_bounds__(256) void qkv_rope_kernel(
    const float* __restrict__ x, const float* __restrict__ Wq,
    const float* __restrict__ Wk, const float* __restrict__ Wv,
    _Float16* __restrict__ qh, _Float16* __restrict__ kh,
    _Float16* __restrict__ vt)
{
    const int tid = blockIdx.x * 256 + threadIdx.x;
    const int p   = tid & 31;        // pair id within token
    const int tok = tid >> 5;        // 0 .. B*S-1
    const int b   = tok / Ss;
    const int s   = tok - b * Ss;
    const int h   = p >> 3;          // head
    const int j   = p & 7;           // pair within head (dims 2j, 2j+1)
    const int de  = h * Dd + 2 * j;
    const int dodd = de + 1;

    const float* xr = x + (size_t)tok * Ee;
    const float qe = dot64(xr, Wq + (size_t)de * Ee);
    const float qo = dot64(xr, Wq + (size_t)dodd * Ee);
    const float ke = dot64(xr, Wk + (size_t)de * Ee);
    const float ko = dot64(xr, Wk + (size_t)dodd * Ee);
    const float ve = dot64(xr, Wv + (size_t)de * Ee);
    const float vo = dot64(xr, Wv + (size_t)dodd * Ee);

    const float inv_freq = __powf(10000.0f, -(float)j * 0.125f);
    float sn, cs;
    __sincosf((float)s * inv_freq, &sn, &cs);
    const float qre = qe * cs - qo * sn, qro = qe * sn + qo * cs;
    const float kre = ke * cs - ko * sn, kro = ke * sn + ko * cs;

    const size_t qb = ((size_t)((b * Hh + h) * Ss) + s) * Dp;
    qh[qb + 2 * j]      = (_Float16)qre;
    qh[qb + 2 * j + 1]  = (_Float16)qro;
    qh[qb + 16 + 2 * j] = (_Float16)0.f;
    qh[qb + 17 + 2 * j] = (_Float16)0.f;
    kh[qb + 2 * j]      = (_Float16)kre;
    kh[qb + 2 * j + 1]  = (_Float16)kro;
    kh[qb + 16 + 2 * j] = (_Float16)0.f;
    kh[qb + 17 + 2 * j] = (_Float16)0.f;

    const size_t vb = (size_t)(b * Hh + h) * Dd;
    vt[(vb + 2 * j)     * Ss + s] = (_Float16)ve;
    vt[(vb + 2 * j + 1) * Ss + s] = (_Float16)vo;
}

__global__ void wo_convert_kernel(const float* __restrict__ Wo,
                                  _Float16* __restrict__ wo16)
{
    const int i = blockIdx.x * 256 + threadIdx.x;
    if (i < Ee * Ee) wo16[i] = (_Float16)Wo[i];
}

// ---------------------------------------------------------------------------
// TDM staging: global -> LDS tiles, tracked with TENSORcnt.
// Descriptor packing per CDNA5 ISA.
// ---------------------------------------------------------------------------
static __device__ __forceinline__ unsigned lds_off(const void* p) {
    return (unsigned)(size_t)p;   // low 32 bits of flat LDS address = LDS offset
}

#if USE_TDM
static __device__ __forceinline__ void tdm_load_2d(
    unsigned lds_addr, const void* gsrc,
    unsigned tensor_d0, unsigned tensor_d1,
    unsigned tile_d0, unsigned tile_d1, unsigned stride0)
{
    unsigned long long ga = (unsigned long long)(size_t)gsrc;
    v4u g0 = { 1u,                                   // count=1 (user mode)
               lds_addr,                             // lds_addr [63:32]
               (unsigned)ga,                         // global_addr lo
               (unsigned)(ga >> 32) | (2u << 30) };  // global_addr hi | type=2
    v8i g1;
    g1[0] = (int)(1u << 16);                                       // data_size=2B
    g1[1] = (int)((tensor_d0 & 0xFFFFu) << 16);                    // dim0 lo16
    g1[2] = (int)((tensor_d0 >> 16) | ((tensor_d1 & 0xFFFFu) << 16));
    g1[3] = (int)((tensor_d1 >> 16) | (tile_d0 << 16));            // tile_dim0
    g1[4] = (int)(tile_d1 & 0xFFFFu);                              // tile_dim1
    g1[5] = (int)stride0;                                          // dim0 stride lo
    g1[6] = 0;
    g1[7] = 0;
    v4i z4 = { 0, 0, 0, 0 };
    v8i z8 = { 0, 0, 0, 0, 0, 0, 0, 0 };
    // 6-arg form (clang-23 / therock-10.0): (g0, g1, g2, g3, g4, cpol)
    __builtin_amdgcn_tensor_load_to_lds(g0, g1, z4, z4, z8, 0);
}
#endif

// stage a 32-key K tile (contiguous 2KB) and a 16x32 V tile (strided rows)
static __device__ __forceinline__ void stage_kv(
    _Float16* kdst, _Float16* vdst,
    const _Float16* ksrc, const _Float16* vsrc, int lane)
{
#if USE_TDM
    (void)lane;
    tdm_load_2d(lds_off(kdst), ksrc, 1024, 1, 1024, 0, 1024);      // 1D 2KB
    tdm_load_2d(lds_off(vdst), vsrc, Ss, Dd, 32, Dd, Ss);          // 16 x 32
#else
#pragma unroll
    for (int i = lane; i < 128; i += 32)               // 1024 halves / 8
        ((v8h*)kdst)[i] = ((const v8h*)ksrc)[i];
#pragma unroll
    for (int i = lane; i < 64; i += 32) {              // 16 rows x 4 v8h
        const int d = i >> 2, c = i & 3;
        *(v8h*)(vdst + d * 32 + c * 8) = *(const v8h*)(vsrc + (size_t)d * Ss + c * 8);
    }
#endif
}

static __device__ __forceinline__ void tensor_wait(int pending) {
#if USE_TDM
    if (pending) __builtin_amdgcn_s_wait_tensorcnt(2);
    else         __builtin_amdgcn_s_wait_tensorcnt(0);
#else
    (void)pending;
#endif
}

// ---------------------------------------------------------------------------
// Per-chunk flash attention step. Softmax done on the A-layout side:
// after one LDS transpose pass, lane owns row M=lane&15 (half of its 32 keys,
// partner half in lane^16) -> in-lane reductions + single xor-16 shuffles.
// ---------------------------------------------------------------------------
template <bool MASKED>
static __device__ __forceinline__ void attn_chunk(
    const _Float16* __restrict__ kb,   // LDS [32 keys][Dp]
    const _Float16* __restrict__ vb,   // LDS [Dd][32 keys]
    float* __restrict__ srow,          // LDS [16][Dp] f32 score staging (per wave)
    const v16h& aq, v8f& o, float& mrow, float& lrow,
    int key0, int qbase, int n, int hiHalf, int Mrow0)
{
    // score WMMAs: keys key0..+15 and key0+16..+31
    const v16h bk0 = cat8(*(const v8h*)(kb + (size_t)n        * Dp + 8 * hiHalf),
                          *(const v8h*)(kb + (size_t)n        * Dp + 16 + 8 * hiHalf));
    const v16h bk1 = cat8(*(const v8h*)(kb + (size_t)(16 + n) * Dp + 8 * hiHalf),
                          *(const v8h*)(kb + (size_t)(16 + n) * Dp + 16 + 8 * hiHalf));
    v8f c0 = {}, c1 = {};
    c0 = wmma_f16(aq, bk0, c0);
    c1 = wmma_f16(aq, bk1, c1);

    // C-layout -> LDS (masked + scaled)
    const int kidx0 = key0 + n, kidx1 = key0 + 16 + n;
#pragma unroll
    for (int r = 0; r < 8; ++r) {
        float s0 = c0[r] * 0.25f, s1 = c1[r] * 0.25f;
        if (MASKED) {
            const int qi = qbase + Mrow0 + r;
            s0 = (kidx0 <= qi) ? s0 : -INFINITY;
            s1 = (kidx1 <= qi) ? s1 : -INFINITY;
        }
        srow[(Mrow0 + r) * Dp + n]      = s0;
        srow[(Mrow0 + r) * Dp + n + 16] = s1;
    }
    __asm__ volatile("" ::: "memory");

    // lane reads its row's 16 values in A-fragment element order
    const float* prow = srow + n * Dp;
    const float4 f0 = *(const float4*)(prow + 8 * hiHalf);
    const float4 f1 = *(const float4*)(prow + 8 * hiHalf + 4);
    const float4 f2 = *(const float4*)(prow + 16 + 8 * hiHalf);
    const float4 f3 = *(const float4*)(prow + 16 + 8 * hiHalf + 4);
    float sv[16] = { f0.x, f0.y, f0.z, f0.w, f1.x, f1.y, f1.z, f1.w,
                     f2.x, f2.y, f2.z, f2.w, f3.x, f3.y, f3.z, f3.w };

    float vmx = sv[0];
#pragma unroll
    for (int e = 1; e < 16; ++e) vmx = fmaxf(vmx, sv[e]);
    vmx = fmaxf(vmx, __shfl_xor(vmx, 16, 32));           // combine partner half
    const float mnew = fmaxf(mrow, vmx);
    const float corr = __expf(mrow - mnew);
    mrow = mnew;

    v16h ap;                                             // P A-fragment, in regs
    float ps = 0.f;
#pragma unroll
    for (int e = 0; e < 16; ++e) {
        const float pe = __expf(sv[e] - mnew);
        ps += pe;
        ap[e] = (_Float16)pe;
    }
    ps += __shfl_xor(ps, 16, 32);
    lrow = lrow * corr + ps;

    // rescale accumulator: row M=r+8*hiHalf is held by lane (M + 16*hiHalf)
#pragma unroll
    for (int r = 0; r < 8; ++r)
        o[r] *= __shfl(corr, r + 24 * hiHalf, 32);

    // P.V accumulate
    const v16h bv = cat8(*(const v8h*)(vb + (size_t)n * Dp + 8 * hiHalf),
                         *(const v8h*)(vb + (size_t)n * Dp + 16 + 8 * hiHalf));
    o = wmma_f16(ap, bv, o);
}

// ---------------------------------------------------------------------------
// Kernel 2: flash attention. 1 wave per 16-query tile, 4 waves per block,
// TDM double-buffered K/V staging.
// ---------------------------------------------------------------------------
__global__ __launch_bounds__(128) void attn_kernel(
    const _Float16* __restrict__ qh, const _Float16* __restrict__ kh,
    const _Float16* __restrict__ vt, _Float16* __restrict__ attn)
{
    __shared__ __attribute__((aligned(64))) _Float16 kbuf[4][2][32 * Dp]; // 16KB
    __shared__ __attribute__((aligned(64))) _Float16 vbuf[4][2][Dd * 32]; //  8KB
    __shared__ __attribute__((aligned(64))) float    sbuf[4][16][Dp];     //  8KB

    const int lane   = threadIdx.x & 31;
    const int wave   = threadIdx.x >> 5;
    const int h      = blockIdx.y;
    const int b      = blockIdx.z;
    const int qbase  = (blockIdx.x * 4 + wave) * 16;
    const int n      = lane & 15;
    const int hiHalf = lane >> 4;
    const int Mrow0  = hiHalf * 8;

    const size_t headBase = (size_t)(b * Hh + h) * Ss;
    const _Float16* kp = kh + headBase * Dp;
    const _Float16* vp = vt + (size_t)(b * Hh + h) * Dd * Ss;

    // Q A-fragment
    v16h aq;
    {
        const _Float16* row = qh + (headBase + qbase + n) * Dp;
        aq = cat8(*(const v8h*)(row + 8 * hiHalf),
                  *(const v8h*)(row + 16 + 8 * hiHalf));
    }

    v8f o = {};
    float mrow = -INFINITY, lrow = 0.f;

    _Float16* kst = &kbuf[wave][0][0];
    _Float16* vst = &vbuf[wave][0][0];
    float*    sst = &sbuf[wave][0][0];

    const int nFull   = qbase >> 5;    // fully-unmasked 32-key chunks
    const int nChunks = nFull + 1;     // + diagonal (masked) chunk

    stage_kv(kst, vst, kp, vp, lane);  // prologue: chunk 0 -> buf 0
    for (int kc = 0; kc < nChunks; ++kc) {
        const int cur = kc & 1;
        if (kc + 1 < nChunks) {
            stage_kv(kst + (cur ^ 1) * (32 * Dp), vst + (cur ^ 1) * (Dd * 32),
                     kp + (size_t)(kc + 1) * 32 * Dp, vp + (kc + 1) * 32, lane);
            tensor_wait(1);            // oldest pair (chunk kc) complete
        } else {
            tensor_wait(0);
        }
        __asm__ volatile("" ::: "memory");

        const int key0 = kc << 5;
        if (kc < nFull)
            attn_chunk<false>(kst + cur * (32 * Dp), vst + cur * (Dd * 32), sst,
                              aq, o, mrow, lrow, key0, qbase, n, hiHalf, Mrow0);
        else
            attn_chunk<true>(kst + cur * (32 * Dp), vst + cur * (Dd * 32), sst,
                             aq, o, mrow, lrow, key0, qbase, n, hiHalf, Mrow0);
    }

    // normalize and write attended (merged heads, f16)
    const float inv = 1.0f / lrow;
#pragma unroll
    for (int r = 0; r < 8; ++r) {
        const float sc = __shfl(inv, r + 24 * hiHalf, 32);
        const int srowi = qbase + Mrow0 + r;
        attn[((size_t)(b * Ss + srowi)) * Ee + h * Dd + n] = (_Float16)(o[r] * sc);
    }
}

// ---------------------------------------------------------------------------
// Kernel 3: out = attended @ Wo^T (f32). 8 WMMAs per 16-row tile.
// ---------------------------------------------------------------------------
__global__ __launch_bounds__(128) void oproj_kernel(
    const _Float16* __restrict__ attn, const _Float16* __restrict__ wo16,
    float* __restrict__ out)
{
    const int lane   = threadIdx.x & 31;
    const int wave   = threadIdx.x >> 5;
    const int row0   = (blockIdx.x * 4 + wave) * 16;
    const int n      = lane & 15;
    const int hiHalf = lane >> 4;

    const _Float16* arow = attn + (size_t)(row0 + n) * Ee;
    v16h a0 = cat8(*(const v8h*)(arow + 8 * hiHalf),
                   *(const v8h*)(arow + 16 + 8 * hiHalf));   // K 0..31
    v16h a1 = cat8(*(const v8h*)(arow + 32 + 8 * hiHalf),
                   *(const v8h*)(arow + 48 + 8 * hiHalf));   // K 32..63

#pragma unroll
    for (int nb = 0; nb < 4; ++nb) {
        const _Float16* wrow = wo16 + (size_t)(nb * 16 + n) * Ee;
        const v16h b0 = cat8(*(const v8h*)(wrow + 8 * hiHalf),
                             *(const v8h*)(wrow + 16 + 8 * hiHalf));
        const v16h b1 = cat8(*(const v8h*)(wrow + 32 + 8 * hiHalf),
                             *(const v8h*)(wrow + 48 + 8 * hiHalf));
        v8f acc = {};
        acc = wmma_f16(a0, b0, acc);
        acc = wmma_f16(a1, b1, acc);
#pragma unroll
        for (int r = 0; r < 8; ++r)
            out[(size_t)(row0 + r + 8 * hiHalf) * Ee + nb * 16 + n] = acc[r];
    }
}

// ---------------------------------------------------------------------------
extern "C" void kernel_launch(void* const* d_in, const int* in_sizes, int n_in,
                              void* d_out, int out_size, void* d_ws, size_t ws_size,
                              hipStream_t stream) {
    (void)in_sizes; (void)n_in; (void)out_size; (void)ws_size;
    const float* x  = (const float*)d_in[0];
    const float* Wq = (const float*)d_in[1];
    const float* Wk = (const float*)d_in[2];
    const float* Wv = (const float*)d_in[3];
    const float* Wo = (const float*)d_in[4];
    float* out = (float*)d_out;

    char* ws = (char*)d_ws;
    _Float16* qh   = (_Float16*)(ws);                      // 4 MB
    _Float16* kh   = (_Float16*)(ws + ((size_t)4  << 20)); // 4 MB
    _Float16* vt   = (_Float16*)(ws + ((size_t)8  << 20)); // 2 MB
    _Float16* attn = (_Float16*)(ws + ((size_t)10 << 20)); // 2 MB
    _Float16* wo16 = (_Float16*)(ws + ((size_t)12 << 20)); // 8 KB

    qkv_rope_kernel<<<(Bb * Ss * 32) / 256, 256, 0, stream>>>(x, Wq, Wk, Wv, qh, kh, vt);
    wo_convert_kernel<<<(Ee * Ee + 255) / 256, 256, 0, stream>>>(Wo, wo16);
    attn_kernel<<<dim3(Ss / 16 / 4, Hh, Bb), 128, 0, stream>>>(qh, kh, vt, attn);
    oproj_kernel<<<(Bb * Ss / 16) / 4, 128, 0, stream>>>(attn, wo16, out);
}
